// SwinBlock_16028817949005
// MI455X (gfx1250) — compile-verified, matrix operand order checked by reading, not run on hardware
//
#include <hip/hip_runtime.h>
#include <hip/hip_bf16.h>

typedef _Float16 v16h __attribute__((ext_vector_type(16)));
typedef float    v8f  __attribute__((ext_vector_type(8)));

// ---------------------------------------------------------------------------
// CDNA5 16-bit A/B fragment layout (wave32): lane&15 = row (A) / col (B src),
// lane>>4 selects the K half-group; per lane the 16 halves are two contiguous
// 16-byte runs (k 0..7 and 16..23 relative) -> ds_load_b128 / global_load_b128.
// All strides/k0 are multiples of 8 halves so the uint4 loads are 16B aligned.
// ---------------------------------------------------------------------------
union FragU { v16h h; uint4 q[2]; };

__device__ __forceinline__ v16h ld_frag(const _Float16* __restrict__ base,
                                        int row0, int stride, int k0, int lane) {
    const int hf = lane >> 4;
    const _Float16* p = base + (row0 + (lane & 15)) * stride + k0 + hf * 8;
    FragU u;
    u.q[0] = *(const uint4*)(p);
    u.q[1] = *(const uint4*)(p + 16);
    return u.h;
}

template <int KS>
__device__ __forceinline__ void ld_afrags(v16h* a, const _Float16* __restrict__ A,
                                          int row0, int stride, int k0, int lane) {
#pragma unroll
    for (int ks = 0; ks < KS; ++ks) a[ks] = ld_frag(A, row0, stride, k0 + ks * 32, lane);
}

template <int KS>
__device__ __forceinline__ v8f wmma_bn(const v16h* a, const _Float16* __restrict__ B,
                                       int row0, int stride, int k0, int lane) {
    v8f acc = {0.f, 0.f, 0.f, 0.f, 0.f, 0.f, 0.f, 0.f};
#pragma unroll
    for (int ks = 0; ks < KS; ++ks) {
        v16h b = ld_frag(B, row0, stride, k0 + ks * 32, lane);
        acc = __builtin_amdgcn_wmma_f32_16x16x32_f16(
                  false, a[ks], false, b, (short)0, acc, false, false);
    }
    return acc;
}

// ---------------------------------------------------------------------------
// fp32 -> f16 weight staging into d_ws (147 KB, L2-resident across all blocks)
//   half-elem offsets: [0)qkv 288*96  [27648)proj 96*96
//                      [36864)fc1 192*96  [55296)fc2 96*192
// ---------------------------------------------------------------------------
__global__ void convert_weights(const float* __restrict__ qkv,
                                const float* __restrict__ proj,
                                const float* __restrict__ fc1,
                                const float* __restrict__ fc2,
                                _Float16* __restrict__ ws) {
    int i = blockIdx.x * 256 + threadIdx.x;
    if (i < 27648)      ws[i] = (_Float16)qkv[i];
    else if (i < 36864) ws[i] = (_Float16)proj[i - 27648];
    else if (i < 55296) ws[i] = (_Float16)fc1[i - 36864];
    else if (i < 73728) ws[i] = (_Float16)fc2[i - 55296];
}

// ---------------------------------------------------------------------------
// One workgroup (8 wave32) per 8x8 window; wave -> (mt = wid&3, nt = wid>>2 + 2j)
// so A fragments are loaded once per stage and reused across the N sweep.
// ---------------------------------------------------------------------------
__global__ __launch_bounds__(256)
void swin_block_kernel(const float* __restrict__ x,
                       const float* __restrict__ qkv_b,
                       const float* __restrict__ proj_b,
                       const float* __restrict__ rpb,
                       const float* __restrict__ n1g, const float* __restrict__ n1b,
                       const float* __restrict__ n2g, const float* __restrict__ n2b,
                       const float* __restrict__ fc1_b,
                       const float* __restrict__ fc2_b,
                       const _Float16* __restrict__ wq,   // [288][96]
                       const _Float16* __restrict__ wp,   // [96][96]
                       const _Float16* __restrict__ w1,   // [192][96]
                       const _Float16* __restrict__ w2,   // [96][192]
                       float* __restrict__ out) {
    // LDS carve, 136704 B total (<=320KB/WG, 2 WGs per WGP):
    //   [0)      XIN  f32 [64][96]   LN1 input / proj output
    //            SC   f32 [64][68]   scores (overlay, temporally disjoint)
    //   [24576)  A16  f16 [64][200]  LN1/LN2 output (K<=96)
    //            AO   f16 [64][104]  attention out (overlay: A16 dead then)
    //   [50176)  QKV  f16 [64][296]  q|k|v, row=token
    //   [88064)  VT   f16 [96][72]   v transposed [head*32+d][token]
    //   [101888) PR   f16 [64][72]   softmax probs (one head)
    //   [111104) HB   f16 [64][200]  MLP hidden (K=192)
    __shared__ __align__(16) unsigned char smem[136704];
    float*    XIN = (float*)(smem);
    float*    SC  = (float*)(smem);
    _Float16* A16 = (_Float16*)(smem + 24576);
    _Float16* AO  = (_Float16*)(smem + 24576);
    _Float16* QKV = (_Float16*)(smem + 50176);
    _Float16* VT  = (_Float16*)(smem + 88064);
    _Float16* PR  = (_Float16*)(smem + 101888);
    _Float16* HB  = (_Float16*)(smem + 111104);

    const int tid  = threadIdx.x;
    const int lane = tid & 31;
    const int wid  = tid >> 5;
    const int mt   = wid & 3;        // this wave's M tile (rows mt*16..mt*16+15)
    const int ng   = wid >> 2;       // N-tile group 0/1
    const int rbase = mt * 16 + ((lane >> 4) << 3);   // D-tile row for element d: rbase + d
    const int ncol  = lane & 15;

    const int w  = blockIdx.x;       // 8192 windows
    const int b  = w >> 10;
    const int wy = (w >> 5) & 31;
    const int wx = w & 31;
    const int y0 = wy << 3, x0 = wx << 3;
    const float* Xb = x   + (size_t)b * 96 * 65536;
    float*       Ob = out + (size_t)b * 96 * 65536;

    // ---- stage window (B,C,H,W -> [token][chan]) ----
    for (int i = tid; i < 96 * 64; i += 256) {
        int c = i >> 6, t = i & 63;
        XIN[t * 96 + c] = Xb[c * 65536 + (y0 + (t >> 3)) * 256 + x0 + (t & 7)];
    }
    __syncthreads();

    // ---- LayerNorm1 -> A16 ----
    if (tid < 64) {
        const float* r = XIN + tid * 96;
        float mu = 0.f;
        for (int c = 0; c < 96; ++c) mu += r[c];
        mu *= (1.f / 96.f);
        float var = 0.f;
        for (int c = 0; c < 96; ++c) { float d = r[c] - mu; var += d * d; }
        float inv = rsqrtf(var * (1.f / 96.f) + 1e-5f);
        _Float16* o = A16 + tid * 200;
        for (int c = 0; c < 96; ++c)
            o[c] = (_Float16)((r[c] - mu) * inv * n1g[c] + n1b[c]);
    }
    __syncthreads();

    // ---- QKV GEMM: [64x96]x[96x288] -> QKV buffer, branch-free epilogue ----
    {
        v16h a[3];
        ld_afrags<3>(a, A16, mt * 16, 200, 0, lane);
        for (int nt = ng; nt < 18; nt += 2) {
            v8f acc = wmma_bn<3>(a, wq, nt * 16, 96, 0, lane);
            int col = nt * 16 + ncol;
            float bias = qkv_b[col];
#pragma unroll
            for (int d = 0; d < 8; ++d)
                QKV[(rbase + d) * 296 + col] = (_Float16)(acc[d] + bias);
        }
    }
    __syncthreads();

    // ---- build transposed V: VT[c][tok] = QKV[tok][192+c] ----
    for (int i = tid; i < 96 * 64; i += 256) {
        int c = i >> 6, t = i & 63;
        VT[c * 72 + t] = QKV[t * 296 + 192 + c];
    }

    // ---- windowed attention, one head at a time ----
    for (int h = 0; h < 3; ++h) {
        __syncthreads();
        // scores = (q k^T)*scale + rel-pos bias : this wave does nt = ng, ng+2
        {
            v16h a[1];
            ld_afrags<1>(a, QKV, mt * 16, 296, h * 32, lane);           // q
            for (int nt = ng; nt < 4; nt += 2) {
                v8f acc = wmma_bn<1>(a, QKV, nt * 16, 296, 96 + h * 32, lane);  // k
                int col = nt * 16 + ncol;
#pragma unroll
                for (int d = 0; d < 8; ++d) {
                    int row = rbase + d;
                    int idx = ((row >> 3) - (col >> 3) + 7) * 15 + ((row & 7) - (col & 7) + 7);
                    SC[row * 68 + col] = acc[d] * 0.17677669529663689f + rpb[idx * 3 + h];
                }
            }
        }
        __syncthreads();
        // softmax over each row of 64
        if (tid < 64) {
            float* r = SC + tid * 68;
            float mx = -3.4e38f;
            for (int j = 0; j < 64; ++j) mx = fmaxf(mx, r[j]);
            float s = 0.f;
            for (int j = 0; j < 64; ++j) { float e = __expf(r[j] - mx); r[j] = e; s += e; }
            float inv = 1.f / s;
            _Float16* o = PR + tid * 72;
            for (int j = 0; j < 64; ++j) o[j] = (_Float16)(r[j] * inv);
        }
        __syncthreads();
        // out_h = P V : 4x2 tiles, one per wave (nt = ng)
        {
            v16h a[2];
            ld_afrags<2>(a, PR, mt * 16, 72, 0, lane);
            v8f acc = wmma_bn<2>(a, VT, h * 32 + ng * 16, 72, 0, lane);
            int col = ng * 16 + ncol;
#pragma unroll
            for (int d = 0; d < 8; ++d)
                AO[(rbase + d) * 104 + h * 32 + col] = (_Float16)acc[d];
        }
    }
    __syncthreads();

    // ---- proj: [64x96]x[96x96] -> f32 into XIN ----
    {
        v16h a[3];
        ld_afrags<3>(a, AO, mt * 16, 104, 0, lane);
        for (int nt = ng; nt < 6; nt += 2) {
            v8f acc = wmma_bn<3>(a, wp, nt * 16, 96, 0, lane);
            int col = nt * 16 + ncol;
            float bias = proj_b[col];
#pragma unroll
            for (int d = 0; d < 8; ++d)
                XIN[(rbase + d) * 96 + col] = acc[d] + bias;
        }
    }
    __syncthreads();

    // ---- LayerNorm2 -> A16 (AO is dead now; same region) ----
    if (tid < 64) {
        const float* r = XIN + tid * 96;
        float mu = 0.f;
        for (int c = 0; c < 96; ++c) mu += r[c];
        mu *= (1.f / 96.f);
        float var = 0.f;
        for (int c = 0; c < 96; ++c) { float d = r[c] - mu; var += d * d; }
        float inv = rsqrtf(var * (1.f / 96.f) + 1e-5f);
        _Float16* o = A16 + tid * 200;
        for (int c = 0; c < 96; ++c)
            o[c] = (_Float16)((r[c] - mu) * inv * n2g[c] + n2b[c]);
    }
    __syncthreads();

    // ---- fc1 + exact GELU: [64x96]x[96x192] -> HB ----
    {
        v16h a[3];
        ld_afrags<3>(a, A16, mt * 16, 200, 0, lane);
        for (int nt = ng; nt < 12; nt += 2) {
            v8f acc = wmma_bn<3>(a, w1, nt * 16, 96, 0, lane);
            int col = nt * 16 + ncol;
            float bias = fc1_b[col];
#pragma unroll
            for (int d = 0; d < 8; ++d) {
                float v = acc[d] + bias;
                v = 0.5f * v * (1.f + erff(v * 0.70710678118654752f));
                HB[(rbase + d) * 200 + col] = (_Float16)v;
            }
        }
    }
    __syncthreads();

    // ---- fc2 + residual -> global (K=192) ----
    {
        v16h a[6];
        ld_afrags<6>(a, HB, mt * 16, 200, 0, lane);
        for (int nt = ng; nt < 6; nt += 2) {
            v8f acc = wmma_bn<6>(a, w2, nt * 16, 192, 0, lane);
            int col = nt * 16 + ncol;
            float bias = fc2_b[col];
#pragma unroll
            for (int d = 0; d < 8; ++d) {
                int row = rbase + d;
                int addr = col * 65536 + (y0 + (row >> 3)) * 256 + x0 + (row & 7);
                Ob[addr] = acc[d] + bias + Xb[addr];
            }
        }
    }
}

extern "C" void kernel_launch(void* const* d_in, const int* in_sizes, int n_in,
                              void* d_out, int out_size, void* d_ws, size_t ws_size,
                              hipStream_t stream) {
    const float* x      = (const float*)d_in[0];
    const float* qkv_w  = (const float*)d_in[1];
    const float* qkv_b  = (const float*)d_in[2];
    const float* proj_w = (const float*)d_in[3];
    const float* proj_b = (const float*)d_in[4];
    const float* rpb    = (const float*)d_in[5];
    const float* n1g    = (const float*)d_in[6];
    const float* n1b    = (const float*)d_in[7];
    const float* n2g    = (const float*)d_in[8];
    const float* n2b    = (const float*)d_in[9];
    const float* fc1_w  = (const float*)d_in[10];
    const float* fc1_b  = (const float*)d_in[11];
    const float* fc2_w  = (const float*)d_in[12];
    const float* fc2_b  = (const float*)d_in[13];
    float* out = (float*)d_out;

    _Float16* ws = (_Float16*)d_ws;
    const _Float16* wqp = ws;
    const _Float16* wpp = ws + 27648;
    const _Float16* w1p = ws + 36864;
    const _Float16* w2p = ws + 55296;

    convert_weights<<<288, 256, 0, stream>>>(qkv_w, proj_w, fc1_w, fc2_w, ws);

    swin_block_kernel<<<8192, 256, 0, stream>>>(
        x, qkv_b, proj_b, rpb, n1g, n1b, n2g, n2b, fc1_b, fc2_b,
        wqp, wpp, w1p, w2p, out);
}
